// GroupQueryAttention_49177375539456
// MI455X (gfx1250) — compile-verified, matrix operand order checked by reading, not run on hardware
//
#include <hip/hip_runtime.h>
#include <hip/hip_bf16.h>

typedef __attribute__((ext_vector_type(16))) _Float16 v16h;
typedef __attribute__((ext_vector_type(8)))  _Float16 v8h;
typedef __attribute__((ext_vector_type(8)))  float    v8f;
typedef __attribute__((address_space(3)))    _Float16 LdsH;

#define DM   2048
#define SQ   2048
#define BB   2
#define NHQ  32
#define NHKV 8
#define HD   64
#define NROW (BB*SQ)   // 4096

union FragH { v16h v; v8h h[2]; };

#define WAIT_ASYNC() asm volatile("s_wait_asynccnt 0x0" ::: "memory")

// ---------------- RMSNorm: x -> xn(f32) + xn(f16) ----------------
__global__ void rmsnorm_kernel(const float* __restrict__ x, const float* __restrict__ w,
                               float* __restrict__ xn, _Float16* __restrict__ xnh) {
  int r = blockIdx.x;
  const float* xr = x + (size_t)r * DM;
  __shared__ float red[256];
  float s = 0.f;
  for (int i = threadIdx.x; i < DM; i += 256) { float v = xr[i]; s += v * v; }
  red[threadIdx.x] = s; __syncthreads();
  for (int off = 128; off > 0; off >>= 1) {
    if (threadIdx.x < off) red[threadIdx.x] += red[threadIdx.x + off];
    __syncthreads();
  }
  float inv = rsqrtf(red[0] / (float)DM + 1.1920929e-7f);
  for (int i = threadIdx.x; i < DM; i += 256) {
    float v = xr[i] * inv * w[i];
    xn [(size_t)r * DM + i] = v;
    xnh[(size_t)r * DM + i] = (_Float16)v;
  }
}

// ---------------- W (K x N, f32) -> Wt (N x K, f16) ----------------
__global__ void wtrans_kernel(const float* __restrict__ W, _Float16* __restrict__ Wt,
                              int K, int N) {
  int idx = blockIdx.x * 256 + threadIdx.x;
  if (idx >= K * N) return;
  int k = idx / N, n = idx % N;
  Wt[(size_t)n * K + k] = (_Float16)W[idx];
}

// ---------------- WMMA GEMM: C[M,N] = A[M,K] @ Bt[N,K]^T ----------------
// Block = 8 waves = 128x64 C tile. Shared 64x32 B tile double-buffered in LDS,
// filled with gfx1250 async-to-LDS copies (ASYNCcnt) overlapping the WMMAs.
template<bool RESID>
__global__ void gemm_wmma(const _Float16* __restrict__ A, const _Float16* __restrict__ Bt,
                          _Float16* __restrict__ Ch, float* __restrict__ Cf,
                          const float* __restrict__ resid, int M, int N, int K) {
  __shared__ __align__(16) _Float16 bs[2][64 * 32];   // 8 KB double buffer
  int tiles_n = N >> 6;
  int bm = blockIdx.x / tiles_n;
  int bn = blockIdx.x % tiles_n;
  int lane = threadIdx.x & 31;
  int wave = threadIdx.x >> 5;
  int tm = bm * 8 + wave;                 // 16-row tile per wave (grid exact)
  int nrow = lane & 15;
  int half = lane >> 4;
  int koff = half * 8;                    // ISA 16-bit A/B lane layout

  // cooperative B-tile fill: 256 threads x 16B = 64x32 halves
  int fn = threadIdx.x >> 2;              // 0..63  (N row of tile)
  int fk = (threadIdx.x & 3) << 3;        // 0,8,16,24 (K offset, halves)
  const _Float16* fsrc_row = Bt + (size_t)(bn * 64 + fn) * K + fk;
  _Float16* fdst0 = &bs[0][fn * 32 + fk];
  _Float16* fdst1 = &bs[1][fn * 32 + fk];

  auto fill = [&](int buf, int k0) {
    const _Float16* src = fsrc_row + k0;
    _Float16* dst = buf ? fdst1 : fdst0;
    // GLOBAL_LOAD_ASYNC_TO_LDS_B128 (GV mode): VDST = LDS byte address (AS3),
    // VADDR = 64-bit global address, no SADDR.  Tracked by ASYNCcnt.
    asm volatile("global_load_async_to_lds_b128 %0, %1, off"
                 :: "v"((LdsH*)dst), "v"(src)
                 : "memory");
  };

  const _Float16* arow = A + (size_t)(tm * 16 + nrow) * K;
  v8f acc[4] = {};
  int buf = 0;
  fill(0, 0);
  for (int k0 = 0; k0 < K; k0 += 32) {
    WAIT_ASYNC();                 // this wave's async chunk landed in LDS
    __syncthreads();              // all waves' chunks visible; prev reads done
    if (k0 + 32 < K) fill(buf ^ 1, k0 + 32);
    FragH a;
    a.h[0] = *(const v8h*)(arow + k0 + koff);
    a.h[1] = *(const v8h*)(arow + k0 + 16 + koff);
    __builtin_prefetch(arow + k0 + 128, 0, 1);    // global_prefetch_b8
#pragma unroll
    for (int t = 0; t < 4; ++t) {
      const _Float16* brow = &bs[buf][(t * 16 + nrow) * 32];
      FragH b;
      b.h[0] = *(const v8h*)(brow + koff);
      b.h[1] = *(const v8h*)(brow + 16 + koff);
      acc[t] = __builtin_amdgcn_wmma_f32_16x16x32_f16(false, a.v, false, b.v,
                                                      (short)0, acc[t], false, false);
    }
    buf ^= 1;
  }
  int rbase = tm * 16 + half * 8;   // C layout: vgpr v -> row v (+8 for upper half)
#pragma unroll
  for (int t = 0; t < 4; ++t) {
    int col = bn * 64 + t * 16 + nrow;
#pragma unroll
    for (int v = 0; v < 8; ++v) {
      size_t idx = (size_t)(rbase + v) * N + col;
      if (RESID) Cf[idx] = acc[t][v] + resid[idx];
      else       Ch[idx] = (_Float16)acc[t][v];
    }
  }
}

// ---------------- RoPE + relayout: (B*S, nh*64) -> (B, nh, S, 64) ----------------
__global__ void rope_kernel(const _Float16* __restrict__ in, _Float16* __restrict__ out,
                            int nh, int total) {
  int idx = blockIdx.x * 256 + threadIdx.x;
  if (idx >= total) return;
  int i = idx & 31;                        // rotation pair index (0..31)
  int h = (idx >> 5) % nh;
  int s = ((idx >> 5) / nh) % SQ;
  int b = idx / (32 * nh * SQ);
  const _Float16* src = in + ((size_t)(b * SQ + s) * nh + h) * HD;
  float x1 = (float)src[i], x2 = (float)src[i + 32];
  float fr = (float)s * __powf(10000.0f, -(float)(2 * i) / (float)HD);
  float c = __cosf(fr), sn = __sinf(fr);
  _Float16* dst = out + (((size_t)b * nh + h) * SQ + s) * HD;
  dst[i]      = (_Float16)(x1 * c - x2 * sn);
  dst[i + 32] = (_Float16)(x2 * c + x1 * sn);
}

// ---------------- V relayout: (B*S, 8*64) -> (B, 8, 64, S) ----------------
__global__ void vtrans_kernel(const _Float16* __restrict__ v, _Float16* __restrict__ vt) {
  int idx = blockIdx.x * 256 + threadIdx.x;
  if (idx >= BB * SQ * NHKV * HD) return;
  int c = idx % (NHKV * HD);
  int s = (idx / (NHKV * HD)) % SQ;
  int b = idx / (NHKV * HD * SQ);
  int h = c >> 6, d = c & 63;
  vt[(((size_t)b * NHKV + h) * HD + d) * SQ + s] = v[idx];
}

// ---------------- Flash attention: one wave per (b, h, 16-query tile) ----------------
__global__ void attn_kernel(const _Float16* __restrict__ qr, const _Float16* __restrict__ kr,
                            const _Float16* __restrict__ vt, _Float16* __restrict__ out,
                            const int* __restrict__ causal_p) {
  __shared__ _Float16 lds_p[8][16 * 32];
  int causal = *causal_p;
  int lane = threadIdx.x & 31;
  int wave = threadIdx.x >> 5;
  int job = blockIdx.x * 8 + wave;          // B * NHQ * (SQ/16) = 8192
  int qt = job % (SQ / 16);
  int h  = (job / (SQ / 16)) % NHQ;
  int b  = job / ((SQ / 16) * NHQ);
  int hk = h >> 2;                          // GROUP = 4
  int q0 = qt * 16;
  int nrow = lane & 15;
  int half = lane >> 4;
  int koff = half * 8;
  const _Float16* kbase = kr + (((size_t)b * NHKV + hk) * SQ) * HD;
  const _Float16* vbase = vt + (((size_t)b * NHKV + hk) * HD) * SQ;
  // Q A-fragments (head dim 64 = two K=32 chunks), loaded once
  const _Float16* qrow = qr + ((((size_t)b * NHQ + h) * SQ) + q0 + nrow) * HD;
  FragH qa[2];
#pragma unroll
  for (int c = 0; c < 2; ++c) {
    qa[c].h[0] = *(const v8h*)(qrow + c * 32 + koff);
    qa[c].h[1] = *(const v8h*)(qrow + c * 32 + 16 + koff);
  }
  float m[8], l[8];
#pragma unroll
  for (int v = 0; v < 8; ++v) { m[v] = -3.0e38f; l[v] = 0.f; }
  v8f o[4] = {};
  int kend = causal ? (q0 + 16) : SQ;
  for (int k0 = 0; k0 < kend; k0 += 32) {
    // ---- scores: 16x32 block as two 16x16 C tiles ----
    v8f sc[2] = {};
#pragma unroll
    for (int t = 0; t < 2; ++t) {
      int krid = k0 + t * 16 + nrow; if (krid > SQ - 1) krid = SQ - 1;  // masked anyway
      const _Float16* krow = kbase + (size_t)krid * HD;
#pragma unroll
      for (int c = 0; c < 2; ++c) {
        FragH kb;
        kb.h[0] = *(const v8h*)(krow + c * 32 + koff);
        kb.h[1] = *(const v8h*)(krow + c * 32 + 16 + koff);
        sc[t] = __builtin_amdgcn_wmma_f32_16x16x32_f16(false, qa[c].v, false, kb.v,
                                                       (short)0, sc[t], false, false);
      }
    }
    // ---- scale + causal mask (C layout: col = lane&15, row = v + 8*half) ----
#pragma unroll
    for (int t = 0; t < 2; ++t)
#pragma unroll
      for (int v = 0; v < 8; ++v) {
        float s = sc[t][v] * 0.125f;   // 1/sqrt(64)
        if (causal && (k0 + t * 16 + nrow > q0 + v + half * 8)) s = -3.0e38f;
        sc[t][v] = s;
      }
    // ---- online softmax; row's 16 cols live in one 16-lane half ----
    float alpha[8];
#pragma unroll
    for (int v = 0; v < 8; ++v) {
      float lm = fmaxf(sc[0][v], sc[1][v]);
#pragma unroll
      for (int off = 8; off > 0; off >>= 1) lm = fmaxf(lm, __shfl_xor(lm, off, 32));
      float mn = fmaxf(m[v], lm);
      alpha[v] = __expf(m[v] - mn);
      m[v] = mn;
    }
    asm volatile("s_wait_dscnt 0x0" ::: "memory");   // WAR vs previous P loads
#pragma unroll
    for (int v = 0; v < 8; ++v) {
      float p0 = __expf(sc[0][v] - m[v]);
      float p1 = __expf(sc[1][v] - m[v]);
      int r = v + half * 8;
      lds_p[wave][r * 32 + nrow]      = (_Float16)p0;
      lds_p[wave][r * 32 + 16 + nrow] = (_Float16)p1;
      float s = p0 + p1;
#pragma unroll
      for (int off = 8; off > 0; off >>= 1) s += __shfl_xor(s, off, 32);
      l[v] = l[v] * alpha[v] + s;
    }
    asm volatile("s_wait_dscnt 0x0" ::: "memory");   // RAW: P stores -> P A-frag load
    // ---- P (16x32) as A-fragment from LDS ----
    const _Float16* prow = &lds_p[wave][nrow * 32];
    FragH pa;
    pa.h[0] = *(const v8h*)(prow + koff);
    pa.h[1] = *(const v8h*)(prow + 16 + koff);
    // ---- O = O*alpha + P @ V  (4 output-dim tiles of 16) ----
#pragma unroll
    for (int t = 0; t < 4; ++t) {
#pragma unroll
      for (int v = 0; v < 8; ++v) o[t][v] *= alpha[v];
      const _Float16* vrow = vbase + (size_t)(t * 16 + nrow) * SQ + k0;
      FragH vb;
      vb.h[0] = *(const v8h*)(vrow + koff);
      vb.h[1] = *(const v8h*)(vrow + 16 + koff);
      o[t] = __builtin_amdgcn_wmma_f32_16x16x32_f16(false, pa.v, false, vb.v,
                                                    (short)0, o[t], false, false);
    }
  }
  // ---- normalize and store to (B, S, HQ*64) f16 ----
#pragma unroll
  for (int t = 0; t < 4; ++t)
#pragma unroll
    for (int v = 0; v < 8; ++v) {
      int q = q0 + v + half * 8;
      float val = o[t][v] / l[v];
      out[((size_t)(b * SQ + q) * NHQ + h) * HD + t * 16 + nrow] = (_Float16)val;
    }
}

extern "C" void kernel_launch(void* const* d_in, const int* in_sizes, int n_in,
                              void* d_out, int out_size, void* d_ws, size_t ws_size,
                              hipStream_t stream) {
  const float* x     = (const float*)d_in[0];
  const float* rms_w = (const float*)d_in[1];
  const float* Wq    = (const float*)d_in[2];
  const float* Wk    = (const float*)d_in[3];
  const float* Wv    = (const float*)d_in[4];
  const float* Wo    = (const float*)d_in[5];
  const int*   causal = (const int*)d_in[6];
  float* out = (float*)d_out;

  char* ws = (char*)d_ws;
  size_t off = 0;
  auto alloc = [&](size_t bytes) -> void* {
    void* p = ws + off; off += (bytes + 255) & ~(size_t)255; return p;
  };
  float*    xn   = (float*)   alloc((size_t)NROW * DM * 4);
  _Float16* xnh  = (_Float16*)alloc((size_t)NROW * DM * 2);
  _Float16* wqt  = (_Float16*)alloc((size_t)DM * DM * 2);
  _Float16* wkt  = (_Float16*)alloc((size_t)(NHKV*HD) * DM * 2);
  _Float16* wvt  = (_Float16*)alloc((size_t)(NHKV*HD) * DM * 2);
  _Float16* wot  = (_Float16*)alloc((size_t)DM * DM * 2);
  _Float16* qh   = (_Float16*)alloc((size_t)NROW * DM * 2);
  _Float16* kh   = (_Float16*)alloc((size_t)NROW * (NHKV*HD) * 2);
  _Float16* vh   = (_Float16*)alloc((size_t)NROW * (NHKV*HD) * 2);
  _Float16* qrot = (_Float16*)alloc((size_t)NROW * DM * 2);
  _Float16* krot = (_Float16*)alloc((size_t)NROW * (NHKV*HD) * 2);
  _Float16* vtb  = (_Float16*)alloc((size_t)NROW * (NHKV*HD) * 2);
  _Float16* atth = (_Float16*)alloc((size_t)NROW * DM * 2);

  rmsnorm_kernel<<<NROW, 256, 0, stream>>>(x, rms_w, xn, xnh);

  wtrans_kernel<<<(DM*DM + 255)/256, 256, 0, stream>>>(Wq, wqt, DM, DM);
  wtrans_kernel<<<(DM*(NHKV*HD) + 255)/256, 256, 0, stream>>>(Wk, wkt, DM, NHKV*HD);
  wtrans_kernel<<<(DM*(NHKV*HD) + 255)/256, 256, 0, stream>>>(Wv, wvt, DM, NHKV*HD);
  wtrans_kernel<<<(DM*DM + 255)/256, 256, 0, stream>>>(Wo, wot, DM, DM);

  // Q/K/V projections: blocks = (M/128)*(N/64)
  gemm_wmma<false><<<(NROW/128)*(DM/64), 256, 0, stream>>>(
      xnh, wqt, qh, nullptr, nullptr, NROW, DM, DM);
  gemm_wmma<false><<<(NROW/128)*((NHKV*HD)/64), 256, 0, stream>>>(
      xnh, wkt, kh, nullptr, nullptr, NROW, NHKV*HD, DM);
  gemm_wmma<false><<<(NROW/128)*((NHKV*HD)/64), 256, 0, stream>>>(
      xnh, wvt, vh, nullptr, nullptr, NROW, NHKV*HD, DM);

  int qtot = BB * SQ * NHQ * 32;
  int ktot = BB * SQ * NHKV * 32;
  rope_kernel<<<(qtot + 255)/256, 256, 0, stream>>>(qh, qrot, NHQ, qtot);
  rope_kernel<<<(ktot + 255)/256, 256, 0, stream>>>(kh, krot, NHKV, ktot);
  vtrans_kernel<<<(BB*SQ*NHKV*HD + 255)/256, 256, 0, stream>>>(vh, vtb);

  attn_kernel<<<BB*NHQ*(SQ/16)/8, 256, 0, stream>>>(qrot, krot, vtb, atth, causal);

  // Output projection + residual (f32 out)
  gemm_wmma<true><<<(NROW/128)*(DM/64), 256, 0, stream>>>(
      atth, wot, nullptr, out, xn, NROW, DM, DM);
}
